// MoEBlock_31834297598404
// MI455X (gfx1250) — matrix operand order repaired and checked by reading, not compiled, not run on hardware
//
#include <hip/hip_runtime.h>
#include <hip/hip_bf16.h>
#include <math.h>

// MoE block: B=4, T=2048, D=1024, E=8, K=2
#define D_DIM   1024
#define E_NUM   8
#define TOKENS  8192      // B*T
#define TOPK    2
#define NSLOTS  16384     // TOKENS * TOPK

#define BM      128       // tokens per block tile
#define BN      128       // output features per block tile
#define KT      32        // K step (bf16 WMMA depth)
#define AP      40        // padded LDS row stride (bf16 elems) to avoid bank conflicts

typedef __attribute__((ext_vector_type(16))) __bf16 v16bf;
typedef __attribute__((ext_vector_type(8)))  __bf16 v8bf;
typedef __attribute__((ext_vector_type(4)))  __bf16 v4bf;
typedef __attribute__((ext_vector_type(8)))  float  v8f;
typedef __attribute__((ext_vector_type(4)))  float  v4f;

// ---------------------------------------------------------------------------
// fp32 -> packed bf16 hi / lo helpers (hi + lo ~= x), whole-vector converts
// ---------------------------------------------------------------------------
static __device__ __forceinline__ v8bf pack_hi8(v4f a, v4f b) {
    v4bf ha = __builtin_convertvector(a, v4bf);
    v4bf hb = __builtin_convertvector(b, v4bf);
    return __builtin_shufflevector(ha, hb, 0, 1, 2, 3, 4, 5, 6, 7);
}
static __device__ __forceinline__ v8bf pack_lo8(v4f a, v4f b) {
    v4bf ha = __builtin_convertvector(a, v4bf);
    v4bf hb = __builtin_convertvector(b, v4bf);
    v4f  da = a - __builtin_convertvector(ha, v4f);
    v4f  db = b - __builtin_convertvector(hb, v4f);
    v4bf la = __builtin_convertvector(da, v4bf);
    v4bf lb = __builtin_convertvector(db, v4bf);
    return __builtin_shufflevector(la, lb, 0, 1, 2, 3, 4, 5, 6, 7);
}

// ---------------------------------------------------------------------------
// Kernel 0: zero the output accumulator and the per-expert counters.
// ---------------------------------------------------------------------------
__global__ void zero_kernel(float* __restrict__ out, int n, int* __restrict__ count) {
    int i = blockIdx.x * blockDim.x + threadIdx.x;
    if (i < n) out[i] = 0.0f;
    if (i < E_NUM) count[i] = 0;
}

// ---------------------------------------------------------------------------
// Kernel 1: gating. One wave32 per token: logits = x . gate_W[e] + gate_b[e],
// top-2 over 8 experts, softmax over the two winning logits.
// ---------------------------------------------------------------------------
__global__ __launch_bounds__(256) void gate_kernel(
    const float* __restrict__ x,        // [TOKENS, D]
    const float* __restrict__ gw,       // [E, D]
    const float* __restrict__ gb,       // [E]
    int*   __restrict__ topIdx,         // [TOKENS, 2]
    float* __restrict__ topW)           // [TOKENS, 2]
{
    const int wave = threadIdx.x >> 5;
    const int lane = threadIdx.x & 31;
    const int t = blockIdx.x * 8 + wave;
    if (t >= TOKENS) return;

    const float* row = x + (size_t)t * D_DIM;
    float acc[E_NUM];
#pragma unroll
    for (int e = 0; e < E_NUM; ++e) acc[e] = 0.0f;

    for (int d = lane; d < D_DIM; d += 32) {
        float xv = row[d];
#pragma unroll
        for (int e = 0; e < E_NUM; ++e)
            acc[e] = fmaf(xv, gw[e * D_DIM + d], acc[e]);
    }
#pragma unroll
    for (int e = 0; e < E_NUM; ++e) {
#pragma unroll
        for (int off = 16; off > 0; off >>= 1)
            acc[e] += __shfl_xor(acc[e], off, 32);
    }

    if (lane == 0) {
        float v[E_NUM];
#pragma unroll
        for (int e = 0; e < E_NUM; ++e) v[e] = acc[e] + gb[e];
        int i0 = 0; float v0 = v[0];
#pragma unroll
        for (int e = 1; e < E_NUM; ++e) if (v[e] > v0) { v0 = v[e]; i0 = e; }
        int i1 = -1; float v1 = -INFINITY;
#pragma unroll
        for (int e = 0; e < E_NUM; ++e)
            if (e != i0 && v[e] > v1) { v1 = v[e]; i1 = e; }
        float tt = __expf(v1 - v0);
        float inv = 1.0f / (1.0f + tt);
        topIdx[2 * t + 0] = i0;
        topIdx[2 * t + 1] = i1;
        topW[2 * t + 0] = inv;
        topW[2 * t + 1] = tt * inv;
    }
}

// ---------------------------------------------------------------------------
// Kernel 2: per-assignment rank within its expert bucket.
// ---------------------------------------------------------------------------
__global__ void rank_kernel(const int* __restrict__ topIdx,
                            int* __restrict__ rank,
                            int* __restrict__ count)
{
    int i = blockIdx.x * blockDim.x + threadIdx.x;
    if (i < NSLOTS) rank[i] = atomicAdd(&count[topIdx[i]], 1);
}

// ---------------------------------------------------------------------------
// Kernel 3: exclusive scan of the 8 expert counts.
// ---------------------------------------------------------------------------
__global__ void scan_kernel(const int* __restrict__ count, int* __restrict__ offset) {
    if (threadIdx.x == 0 && blockIdx.x == 0) {
        int run = 0;
        for (int e = 0; e < E_NUM; ++e) { offset[e] = run; run += count[e]; }
    }
}

// ---------------------------------------------------------------------------
// Kernel 4: scatter assignments into per-expert compact lists.
// ---------------------------------------------------------------------------
__global__ void scatter_kernel(const int* __restrict__ topIdx,
                               const float* __restrict__ topW,
                               const int* __restrict__ rank,
                               const int* __restrict__ offset,
                               int*   __restrict__ tokIds,
                               float* __restrict__ tokWgt)
{
    int i = blockIdx.x * blockDim.x + threadIdx.x;
    if (i < NSLOTS) {
        int slot = offset[topIdx[i]] + rank[i];
        tokIds[slot] = i >> 1;       // token id
        tokWgt[slot] = topW[i];      // softmax weight
    }
}

// ---------------------------------------------------------------------------
// Kernel 5: expert GEMM, bf16 split-precision WMMA with cooperative LDS
// staging of pre-converted hi/lo planes.
//   grid  = (D/BN n-tiles, max m-tiles, E), block = 256 threads = 8 waves.
//   Block tile: BM=128 gathered tokens x BN=128 features, K step = 32.
//   Per K step: all threads convert A(128x32) + B(128x32) fp32 -> bf16 hi/lo
//   in LDS once; each wave then owns one 16-row strip and 8 N sub-tiles:
//   24 WMMAs per wave per K step. Next K-step global loads are issued before
//   the compute stage (2-stage register pipeline) to hide VMEM latency.
// ---------------------------------------------------------------------------
__global__ __launch_bounds__(256) void moe_gemm_kernel(
    const float* __restrict__ delta,    // [TOKENS, D]
    const float* __restrict__ expertW,  // [E, D, D]
    const float* __restrict__ expertB,  // [E, D]
    const int*   __restrict__ count,    // [E]
    const int*   __restrict__ offset,   // [E]
    const int*   __restrict__ tokIds,   // [NSLOTS]
    const float* __restrict__ tokWgt,   // [NSLOTS]
    float* __restrict__ out)            // [TOKENS, D]
{
    const int e   = blockIdx.z;
    const int cnt = count[e];
    const int mBase = blockIdx.y * BM;
    if (mBase >= cnt) return;                 // uniform early exit
    const int nBase = blockIdx.x * BN;
    const int seg   = offset[e];

    const int tid  = threadIdx.x;
    const int wave = tid >> 5;                // 0..7
    const int lane = tid & 31;
    const int half = lane >> 4;               // 0/1
    const int r    = lane & 15;               // 0..15

    __shared__ __attribute__((aligned(16))) __bf16 sAhi[BM][AP];
    __shared__ __attribute__((aligned(16))) __bf16 sAlo[BM][AP];
    __shared__ __attribute__((aligned(16))) __bf16 sBhi[BN][AP];
    __shared__ __attribute__((aligned(16))) __bf16 sBlo[BN][AP];

    // ---- fill-stage assignments: 2 threads per row, 16 floats each ----
    const int fRow = tid >> 1;                // 0..127 (A and B rows alike)
    const int fK0  = (tid & 1) * 16;          // 0 or 16
    int aSlot = mBase + fRow;
    if (aSlot >= cnt) aSlot = cnt - 1;        // clamp ragged tail
    const float* aSrc = delta + (size_t)tokIds[seg + aSlot] * D_DIM + fK0;
    const float* bSrc = expertW + ((size_t)e * D_DIM + (nBase + fRow)) * D_DIM + fK0;

    // ---- compute-stage assignments ----
    const int cRow = (wave << 4) + r;         // A row in LDS for this lane

    v8f c[8];
#pragma unroll
    for (int nt = 0; nt < 8; ++nt) c[nt] = (v8f){};

    // prologue: load K step 0 into registers
    v4f aReg0 = *(const v4f*)(aSrc);
    v4f aReg1 = *(const v4f*)(aSrc + 4);
    v4f aReg2 = *(const v4f*)(aSrc + 8);
    v4f aReg3 = *(const v4f*)(aSrc + 12);
    v4f bReg0 = *(const v4f*)(bSrc);
    v4f bReg1 = *(const v4f*)(bSrc + 4);
    v4f bReg2 = *(const v4f*)(bSrc + 8);
    v4f bReg3 = *(const v4f*)(bSrc + 12);

    for (int kk = 0; kk < D_DIM; kk += KT) {
        // ---- stage 1: convert current registers into LDS planes ----
        *(v8bf*)&sAhi[fRow][fK0]     = pack_hi8(aReg0, aReg1);
        *(v8bf*)&sAhi[fRow][fK0 + 8] = pack_hi8(aReg2, aReg3);
        *(v8bf*)&sAlo[fRow][fK0]     = pack_lo8(aReg0, aReg1);
        *(v8bf*)&sAlo[fRow][fK0 + 8] = pack_lo8(aReg2, aReg3);
        *(v8bf*)&sBhi[fRow][fK0]     = pack_hi8(bReg0, bReg1);
        *(v8bf*)&sBhi[fRow][fK0 + 8] = pack_hi8(bReg2, bReg3);
        *(v8bf*)&sBlo[fRow][fK0]     = pack_lo8(bReg0, bReg1);
        *(v8bf*)&sBlo[fRow][fK0 + 8] = pack_lo8(bReg2, bReg3);
        __syncthreads();

        // ---- issue next K-step global loads (latency hidden by compute) ----
        if (kk + KT < D_DIM) {
            aReg0 = *(const v4f*)(aSrc + kk + KT);
            aReg1 = *(const v4f*)(aSrc + kk + KT + 4);
            aReg2 = *(const v4f*)(aSrc + kk + KT + 8);
            aReg3 = *(const v4f*)(aSrc + kk + KT + 12);
            bReg0 = *(const v4f*)(bSrc + kk + KT);
            bReg1 = *(const v4f*)(bSrc + kk + KT + 4);
            bReg2 = *(const v4f*)(bSrc + kk + KT + 8);
            bReg3 = *(const v4f*)(bSrc + kk + KT + 12);
        }

        // ---- stage 2: fragments from LDS + 24 WMMAs ----
        // A fragment: lanes 0-15 rows, half selects K {0..7,16..23} vs {8..15,24..31}
        v8bf a0 = *(const v8bf*)&sAhi[cRow][8 * half];
        v8bf a1 = *(const v8bf*)&sAhi[cRow][16 + 8 * half];
        v8bf a2 = *(const v8bf*)&sAlo[cRow][8 * half];
        v8bf a3 = *(const v8bf*)&sAlo[cRow][16 + 8 * half];
        v16bf aHi = __builtin_shufflevector(a0, a1, 0,1,2,3,4,5,6,7,8,9,10,11,12,13,14,15);
        v16bf aLo = __builtin_shufflevector(a2, a3, 0,1,2,3,4,5,6,7,8,9,10,11,12,13,14,15);

#pragma unroll
        for (int nt = 0; nt < 8; ++nt) {
            const int f = nt * 16 + r;   // B row (output feature within tile)
            v8bf b0 = *(const v8bf*)&sBhi[f][8 * half];
            v8bf b1 = *(const v8bf*)&sBhi[f][16 + 8 * half];
            v8bf b2 = *(const v8bf*)&sBlo[f][8 * half];
            v8bf b3 = *(const v8bf*)&sBlo[f][16 + 8 * half];
            v16bf bHi = __builtin_shufflevector(b0, b1, 0,1,2,3,4,5,6,7,8,9,10,11,12,13,14,15);
            v16bf bLo = __builtin_shufflevector(b2, b3, 0,1,2,3,4,5,6,7,8,9,10,11,12,13,14,15);

            // fp32 ~= hi*hi + hi*lo + lo*hi, accumulated in f32
            c[nt] = __builtin_amdgcn_wmma_f32_16x16x32_bf16(false, aHi, false, bHi,
                                                            (short)0, c[nt], false, false);
            c[nt] = __builtin_amdgcn_wmma_f32_16x16x32_bf16(false, aHi, false, bLo,
                                                            (short)0, c[nt], false, false);
            c[nt] = __builtin_amdgcn_wmma_f32_16x16x32_bf16(false, aLo, false, bHi,
                                                            (short)0, c[nt], false, false);
        }
        __syncthreads();
    }

    // ---- weighted scatter-add into the output ----
    // C tile layout: lane = col (r), VGPR j = row j + 8*half within the
    // wave's 16-row strip.
    int   rowTok[8];
    float rowW[8];
    bool  rowOk[8];
#pragma unroll
    for (int j = 0; j < 8; ++j) {
        int slot = mBase + (wave << 4) + j + 8 * half;
        rowOk[j] = (slot < cnt);
        int s = rowOk[j] ? slot : (cnt - 1);
        rowTok[j] = tokIds[seg + s];
        rowW[j]   = tokWgt[seg + s];
    }
#pragma unroll
    for (int nt = 0; nt < 8; ++nt) {
        const int col = nBase + nt * 16 + r;
        const float bias = expertB[(size_t)e * D_DIM + col];
#pragma unroll
        for (int j = 0; j < 8; ++j) {
            if (rowOk[j]) {
                atomicAdd(&out[(size_t)rowTok[j] * D_DIM + col],
                          rowW[j] * (c[nt][j] + bias));
            }
        }
    }
}

// ---------------------------------------------------------------------------
// Host-side launcher
// ---------------------------------------------------------------------------
extern "C" void kernel_launch(void* const* d_in, const int* in_sizes, int n_in,
                              void* d_out, int out_size, void* d_ws, size_t ws_size,
                              hipStream_t stream) {
    (void)in_sizes; (void)n_in; (void)ws_size;

    const float* input_feat = (const float*)d_in[0];   // [B,T,D]
    const float* delta      = (const float*)d_in[1];   // [B,T,D]
    const float* gate_W     = (const float*)d_in[2];   // [E,D]
    const float* gate_b     = (const float*)d_in[3];   // [E]
    const float* expert_W   = (const float*)d_in[4];   // [E,D,D]
    const float* expert_b   = (const float*)d_in[5];   // [E,D]
    float* out = (float*)d_out;

    // workspace carve-up (256B aligned slices)
    char* ws = (char*)d_ws;
    auto carve = [&ws](size_t bytes) { char* p = ws; ws += (bytes + 255) & ~(size_t)255; return p; };
    int*   topIdx = (int*)  carve(NSLOTS * sizeof(int));
    float* topW   = (float*)carve(NSLOTS * sizeof(float));
    int*   rank   = (int*)  carve(NSLOTS * sizeof(int));
    int*   tokIds = (int*)  carve(NSLOTS * sizeof(int));
    float* tokWgt = (float*)carve(NSLOTS * sizeof(float));
    int*   count  = (int*)  carve(E_NUM * sizeof(int));
    int*   offset = (int*)  carve(E_NUM * sizeof(int));

    // 0) zero output + counters
    {
        int n = out_size;
        int blocks = (n + 255) / 256;
        zero_kernel<<<blocks, 256, 0, stream>>>(out, n, count);
    }
    // 1) gating
    gate_kernel<<<TOKENS / 8, 256, 0, stream>>>(input_feat, gate_W, gate_b, topIdx, topW);
    // 2) ranks within expert buckets
    rank_kernel<<<(NSLOTS + 255) / 256, 256, 0, stream>>>(topIdx, rank, count);
    // 3) expert offsets
    scan_kernel<<<1, 32, 0, stream>>>(count, offset);
    // 4) compact per-expert token lists
    scatter_kernel<<<(NSLOTS + 255) / 256, 256, 0, stream>>>(topIdx, topW, rank, offset,
                                                             tokIds, tokWgt);
    // 5) expert GEMM + weighted combine
    {
        dim3 grid(D_DIM / BN, (NSLOTS + BM - 1) / BM, E_NUM);
        moe_gemm_kernel<<<grid, 256, 0, stream>>>(delta, expert_W, expert_b,
                                                  count, offset, tokIds, tokWgt, out);
    }
}
